// LMKANRBFLayer_60722247631487
// MI455X (gfx1250) — matrix operand order repaired and checked by reading, not compiled, not run on hardware
//
#include <hip/hip_runtime.h>
#include <hip/hip_bf16.h>
#include <math.h>

typedef _Float16 v8h  __attribute__((ext_vector_type(8)));
typedef _Float16 v16h __attribute__((ext_vector_type(16)));
typedef float    v8f  __attribute__((ext_vector_type(8)));

#define IN_DIM  512
#define OUT_DIM 1024
#define KC      16
#define HID     8
#define BATCH   8192
#define FDIM    (IN_DIM * KC)   /* 8192 feature columns (phi part) */

/* ---------------- workspace layout (bytes) ----------------
   phi   : [BATCH][FDIM]  f16   @ 0          (128 MB)
   WhT   : [OUT_DIM][FDIM] f16  @ 134217728  ( 16 MB)
   logdet: [BATCH] f32          @ 150994944  ( 32 KB)
   invw  : [FDIM]  f32          @ 151027712  ( 32 KB)
---------------------------------------------------------- */
#define WS_PHI_OFF    0
#define WS_WHT_OFF    134217728UL
#define WS_LOGDET_OFF 150994944UL
#define WS_INVW_OFF   151027712UL

// ---------------------------------------------------------------------------
// 1) invw[i] = 1 / (exp(log_widths[i]) + 1e-12)
// ---------------------------------------------------------------------------
__global__ void prep_invw_kernel(const float* __restrict__ lw,
                                 float* __restrict__ invw) {
  int i = blockIdx.x * 256 + threadIdx.x;
  if (i < FDIM) invw[i] = 1.0f / (__expf(lw[i]) + 1e-12f);
}

// ---------------------------------------------------------------------------
// 2) WhT[n][k] = (f16) Wout[k][n]   for k < FDIM  (LDS-tiled transpose)
// ---------------------------------------------------------------------------
__global__ void __launch_bounds__(256)
prep_wout_kernel(const float* __restrict__ Wout, _Float16* __restrict__ WhT) {
  __shared__ _Float16 tile[32][33];
  const int n0 = blockIdx.x * 32;
  const int k0 = blockIdx.y * 32;
  const int t  = threadIdx.x;
  {
    int n = t & 31, kk = t >> 5;           // coalesced reads along n
#pragma unroll
    for (int p = 0; p < 4; ++p) {
      int k = kk + p * 8;
      tile[k][n] = (_Float16)Wout[(size_t)(k0 + k) * OUT_DIM + n0 + n];
    }
  }
  __syncthreads();
  {
    int k = t & 31, nn = t >> 5;           // coalesced writes along k
#pragma unroll
    for (int p = 0; p < 4; ++p) {
      int n = nn + p * 8;
      WhT[(size_t)(n0 + n) * FDIM + k0 + k] = tile[k][n];
    }
  }
}

// ---------------------------------------------------------------------------
// 3) metric net + RBF features.  One block (512 threads) per batch row.
//    phi[row][d*16+k] = exp(-2*((x_d*sqrt(g_d) - c_dk) * invw_dk)^2)  (f16)
//    logdet[row] = sum_d log(g_d)
// ---------------------------------------------------------------------------
__global__ void __launch_bounds__(512)
metric_phi_kernel(const float* __restrict__ x,
                  const float* __restrict__ W1, const float* __restrict__ b1,
                  const float* __restrict__ W2, const float* __restrict__ b2,
                  const float* __restrict__ W3, const float* __restrict__ b3,
                  const float* __restrict__ centers,
                  const float* __restrict__ invw,
                  _Float16* __restrict__ phi, float* __restrict__ logdet) {
  __shared__ float s_part[16][HID];
  __shared__ float s_h1[HID];
  __shared__ float s_h2[HID];
  __shared__ float s_ld;

  const int row  = blockIdx.x;
  const int tid  = threadIdx.x;          // == input dim d
  const int lane = tid & 31;
  const int wave = tid >> 5;
  if (tid == 0) s_ld = 0.0f;

  const float xv = x[(size_t)row * IN_DIM + tid];

  // ---- h1 = silu(x @ W1 + b1): per-thread partials, wave shfl-reduce ----
  float p[HID];
#pragma unroll
  for (int h = 0; h < HID; ++h) p[h] = xv * W1[tid * HID + h];
#pragma unroll
  for (int off = 16; off > 0; off >>= 1)
#pragma unroll
    for (int h = 0; h < HID; ++h) p[h] += __shfl_down(p[h], off, 32);
  if (lane == 0)
#pragma unroll
    for (int h = 0; h < HID; ++h) s_part[wave][h] = p[h];
  __syncthreads();

  if (tid < HID) {
    float a = b1[tid];
#pragma unroll
    for (int w = 0; w < 16; ++w) a += s_part[w][tid];
    s_h1[tid] = a / (1.0f + __expf(-a));          // silu
  }
  __syncthreads();
  if (tid < HID) {
    float a = b2[tid];
#pragma unroll
    for (int j = 0; j < HID; ++j) a += s_h1[j] * W2[j * HID + tid];
    s_h2[tid] = a / (1.0f + __expf(-a));          // silu
  }
  __syncthreads();

  // ---- g_d = softplus(h2 @ W3 + b3) + eps, per thread ----
  float s = b3[tid];
#pragma unroll
  for (int j = 0; j < HID; ++j) s += s_h2[j] * W3[j * IN_DIM + tid];
  const float sp = fmaxf(s, 0.0f) + log1pf(__expf(-fabsf(s)));  // stable softplus
  const float g  = sp + 1e-8f;
  const float z  = xv * __fsqrt_rn(g);

  // ---- logdet reduction: wave shfl + one LDS float atomic per wave ----
  float lg = __logf(g);
#pragma unroll
  for (int off = 16; off > 0; off >>= 1) lg += __shfl_down(lg, off, 32);
  if (lane == 0) atomicAdd(&s_ld, lg);            // ds_add_f32

  // ---- 16 RBFs, two b128 f16 stores ----
  const float* cp = centers + tid * KC;
  const float* wp = invw + tid * KC;
  v8h o0, o1;
#pragma unroll
  for (int k = 0; k < 8; ++k) {
    float u = (z - cp[k]) * wp[k];
    o0[k] = (_Float16)__expf(-2.0f * u * u);      // GAMMA = 2
  }
#pragma unroll
  for (int k = 8; k < 16; ++k) {
    float u = (z - cp[k]) * wp[k];
    o1[k - 8] = (_Float16)__expf(-2.0f * u * u);
  }
  v8h* dst = (v8h*)(phi + (size_t)row * FDIM + tid * KC);
  dst[0] = o0;
  dst[1] = o1;

  __syncthreads();
  if (tid == 0) logdet[row] = s_ld;
}

// ---------------------------------------------------------------------------
// A-fragment loader (16-bit A 16x32 layout): two b128 loads -> v16h
// ---------------------------------------------------------------------------
__device__ __forceinline__ v16h load_a16(const _Float16* p) {
  v8h lo = *(const v8h*)p;
  v8h hi = *(const v8h*)(p + 8);
  return __builtin_shufflevector(lo, hi,
      0,1,2,3,4,5,6,7,8,9,10,11,12,13,14,15);
}

#define WMMA_F16(A, B, C) \
  __builtin_amdgcn_wmma_f32_16x16x32_f16(false, (A), false, (B), (short)0, (C), false, false)

#define WMMA_8X(A0, A1, B0, B1, B2, B3)            \
  do {                                             \
    acc[0][0] = WMMA_F16((A0), (B0), acc[0][0]);   \
    acc[1][0] = WMMA_F16((A1), (B0), acc[1][0]);   \
    acc[0][1] = WMMA_F16((A0), (B1), acc[0][1]);   \
    acc[1][1] = WMMA_F16((A1), (B1), acc[1][1]);   \
    acc[0][2] = WMMA_F16((A0), (B2), acc[0][2]);   \
    acc[1][2] = WMMA_F16((A1), (B2), acc[1][2]);   \
    acc[0][3] = WMMA_F16((A0), (B3), acc[0][3]);   \
    acc[1][3] = WMMA_F16((A1), (B3), acc[1][3]);   \
  } while (0)

// ---------------------------------------------------------------------------
// 4) out[m][n] = phi[m][:] @ WhT[n][:] + logdet[m]*Wout[FDIM][n] + bout[n]
//    8 waves/block; wave tile 32(M) x 64(N) => block tile 128x128.
//    Manually ping-ponged double-buffered K-loop (no full unroll),
//    v_wmma_f32_16x16x32_f16, f32 accumulate.
// ---------------------------------------------------------------------------
__global__ void __launch_bounds__(256, 1)
gemm_wmma_kernel(const _Float16* __restrict__ A,   // phi  [BATCH][FDIM]
                 const _Float16* __restrict__ BT,  // WhT  [OUT_DIM][FDIM]
                 const float* __restrict__ logdet,
                 const float* __restrict__ Wout,   // full [FDIM+1][OUT_DIM]
                 const float* __restrict__ bout,
                 float* __restrict__ out) {
  const int lane = threadIdx.x & 31;
  const int wave = threadIdx.x >> 5;
  const int wm = wave & 3;            // 4 waves along M
  const int wn = wave >> 2;           // 2 waves along N
  const int m0 = blockIdx.y * 128 + wm * 32;
  const int n0 = blockIdx.x * 128 + wn * 64;

  // A fragment addressing (contiguous k order, same as B):
  //   lane M = lane&15;  lanes 0-15 carry K 0..15, lanes 16-31 K 16..31
  const int am   = lane & 15;
  const int koff = (lane >> 4) * 16;
  const _Float16* a0p = A + (size_t)(m0 + am) * FDIM + koff;
  const _Float16* a1p = a0p + (size_t)16 * FDIM;

  // B fragment addressing (contiguous k order, same as A):
  //   lane N = lane&15;  lanes 0-15 K=0..15, lanes 16-31 K=16..31 (contiguous in WhT)
  const int bn = lane & 15;
  const int kb = (lane >> 4) * 16;
  const _Float16* bp0 = BT + (size_t)(n0 +  0 + bn) * FDIM + kb;
  const _Float16* bp1 = BT + (size_t)(n0 + 16 + bn) * FDIM + kb;
  const _Float16* bp2 = BT + (size_t)(n0 + 32 + bn) * FDIM + kb;
  const _Float16* bp3 = BT + (size_t)(n0 + 48 + bn) * FDIM + kb;

  v8f acc[2][4];
#pragma unroll
  for (int i = 0; i < 2; ++i)
#pragma unroll
    for (int j = 0; j < 4; ++j) acc[i][j] = (v8f)0.0f;

  // ---- prologue: "cur" buffer = fragments at k = 0 ----
  v16h ca0 = load_a16(a0p);
  v16h ca1 = load_a16(a1p);
  v16h cb0 = *(const v16h*)bp0;
  v16h cb1 = *(const v16h*)bp1;
  v16h cb2 = *(const v16h*)bp2;
  v16h cb3 = *(const v16h*)bp3;
  v16h na0, na1, nb0, nb1, nb2, nb3;

  // ---- ping-pong main loop: two K-steps per iteration, NOT unrolled ----
#pragma unroll 1
  for (int k = 0; k < FDIM - 64; k += 64) {
    // near-cache prefetch ~1KB (16 K-steps) ahead
    __builtin_prefetch(a0p + k + 512, 0, 3);
    __builtin_prefetch(a1p + k + 512, 0, 3);
    __builtin_prefetch(bp0 + k + 512, 0, 3);
    __builtin_prefetch(bp2 + k + 512, 0, 3);

    // phase 1: load k+32 into "next", compute k from "cur"
    na0 = load_a16(a0p + k + 32);
    na1 = load_a16(a1p + k + 32);
    nb0 = *(const v16h*)(bp0 + k + 32);
    nb1 = *(const v16h*)(bp1 + k + 32);
    nb2 = *(const v16h*)(bp2 + k + 32);
    nb3 = *(const v16h*)(bp3 + k + 32);
    WMMA_8X(ca0, ca1, cb0, cb1, cb2, cb3);

    // phase 2: load k+64 into "cur", compute k+32 from "next"
    ca0 = load_a16(a0p + k + 64);
    ca1 = load_a16(a1p + k + 64);
    cb0 = *(const v16h*)(bp0 + k + 64);
    cb1 = *(const v16h*)(bp1 + k + 64);
    cb2 = *(const v16h*)(bp2 + k + 64);
    cb3 = *(const v16h*)(bp3 + k + 64);
    WMMA_8X(na0, na1, nb0, nb1, nb2, nb3);
  }

  // ---- tail: "cur" holds k = FDIM-64; last fragment is k = FDIM-32 ----
  na0 = load_a16(a0p + FDIM - 32);
  na1 = load_a16(a1p + FDIM - 32);
  nb0 = *(const v16h*)(bp0 + FDIM - 32);
  nb1 = *(const v16h*)(bp1 + FDIM - 32);
  nb2 = *(const v16h*)(bp2 + FDIM - 32);
  nb3 = *(const v16h*)(bp3 + FDIM - 32);
  WMMA_8X(ca0, ca1, cb0, cb1, cb2, cb3);
  WMMA_8X(na0, na1, nb0, nb1, nb2, nb3);

  // ---- epilogue: fused logdet column + bias ----
  // C/D layout: VGPR r, lanes 0-15 -> (M=r, N=lane); lanes 16-31 -> (M=r+8, N=lane-16)
  const float* wlast = Wout + (size_t)FDIM * OUT_DIM;  // row 8192 of [8193,1024]
  const int rowoff = (lane >> 4) * 8;
  const int nc = lane & 15;
#pragma unroll
  for (int mt = 0; mt < 2; ++mt) {
#pragma unroll
    for (int nt = 0; nt < 4; ++nt) {
      const int n = n0 + nt * 16 + nc;
      const float wl = wlast[n];
      const float bo = bout[n];
#pragma unroll
      for (int r = 0; r < 8; ++r) {
        const int m = m0 + mt * 16 + rowoff + r;
        out[(size_t)m * OUT_DIM + n] = acc[mt][nt][r] + logdet[m] * wl + bo;
      }
    }
  }
}

// ---------------------------------------------------------------------------
extern "C" void kernel_launch(void* const* d_in, const int* in_sizes, int n_in,
                              void* d_out, int out_size, void* d_ws, size_t ws_size,
                              hipStream_t stream) {
  const float* x          = (const float*)d_in[0];
  const float* W1         = (const float*)d_in[1];
  const float* b1         = (const float*)d_in[2];
  const float* W2         = (const float*)d_in[3];
  const float* b2         = (const float*)d_in[4];
  const float* W3         = (const float*)d_in[5];
  const float* b3         = (const float*)d_in[6];
  const float* centers    = (const float*)d_in[7];
  const float* log_widths = (const float*)d_in[8];
  const float* Wout       = (const float*)d_in[9];
  const float* bout       = (const float*)d_in[10];

  char* ws = (char*)d_ws;
  _Float16* phi  = (_Float16*)(ws + WS_PHI_OFF);
  _Float16* WhT  = (_Float16*)(ws + WS_WHT_OFF);
  float* logdet  = (float*)(ws + WS_LOGDET_OFF);
  float* invw    = (float*)(ws + WS_INVW_OFF);

  prep_invw_kernel<<<FDIM / 256, 256, 0, stream>>>(log_widths, invw);
  prep_wout_kernel<<<dim3(OUT_DIM / 32, FDIM / 32), 256, 0, stream>>>(Wout, WhT);
  metric_phi_kernel<<<BATCH, 512, 0, stream>>>(x, W1, b1, W2, b2, W3, b3,
                                               centers, invw, phi, logdet);
  gemm_wmma_kernel<<<dim3(OUT_DIM / 128, BATCH / 128), 256, 0, stream>>>(
      phi, WhT, logdet, Wout, bout, (float*)d_out);
}
